// IoUMetricLoss_23553600651748
// MI455X (gfx1250) — compile-verified, matrix operand order checked by reading, not run on hardware
//
#include <hip/hip_runtime.h>

// ---------------------------------------------------------------------------
// IoU metric loss for MI455X (gfx1250, wave32).
//
// Stage 1 (bandwidth-bound, roofline ~15us @ 23.3 TB/s for 352 MB):
//   per-pixel argmax over 19 fp32 class planes + per-block LDS histograms.
//   All traffic via b128 loads: 2x float4 per class step, 4x int4 for labels.
//   Partial histograms written as exact f16 counts (<= 2048 per bin).
// Stage 2 (tiny, L2-resident): WMMA tree reduction of 2048 partial histograms
//   with v_wmma_f32_16x16x32_f16 and an all-ones A matrix (each D row =
//   column sums of B), exact f32 accumulation, then the IoU/nanmean scalar.
// ---------------------------------------------------------------------------

#define NUM_CLASSES 19
#define LOG2_HW     19
#define HW          (512 * 1024)          // 2^19
#define HW4         (HW / 4)
#define BATCH       8
#define NPIX        (BATCH * HW)          // 4,194,304
#define PIX_PER_BLK 2048                  // => per-bin count <= 2048 (exact in f16)
#define NBLK        (NPIX / PIX_PER_BLK)  // 2048
#define HIST_STRIDE 96                    // 3 histograms x 32 padded bins

typedef __attribute__((ext_vector_type(16))) _Float16 v16h;
typedef __attribute__((ext_vector_type(8)))  float    v8f;

// ---------------------------------------------------------------------------
// Stage 1: argmax + per-block histogram.
// Thread t of block blk owns 8 consecutive pixels starting at blk*2048 + t*8.
// A block never straddles a batch image (2048 | 2^19), so the batch index is
// uniform per thread and all vector loads are naturally aligned.
// ---------------------------------------------------------------------------
__global__ __launch_bounds__(256) void iou_hist_kernel(
    const float* __restrict__ pred,
    const long long* __restrict__ label,
    _Float16* __restrict__ partial)
{
    __shared__ unsigned int hist[HIST_STRIDE];

    const int t = threadIdx.x;
    for (int k = t; k < HIST_STRIDE; k += 256) hist[k] = 0u;
    __syncthreads();

    const int blk = blockIdx.x;
    const int p0  = blk * PIX_PER_BLK + t * 8;   // first of 8 consecutive pixels
    const int b   = p0 >> LOG2_HW;               // batch image (uniform per thread)
    const int hw  = p0 & (HW - 1);               // multiple of 8 -> float4 aligned

    // float4 view of this image's class planes, at this thread's pixel group.
    const float4* __restrict__ pp =
        (const float4*)pred + (size_t)b * (size_t)(NUM_CLASSES * HW4) + (size_t)(hw >> 2);

    float mx[8];
    int   am[8];

    // class 0
    {
        const float4 c0 = pp[0];
        const float4 c1 = pp[1];
        mx[0] = c0.x; mx[1] = c0.y; mx[2] = c0.z; mx[3] = c0.w;
        mx[4] = c1.x; mx[5] = c1.y; mx[6] = c1.z; mx[7] = c1.w;
#pragma unroll
        for (int i = 0; i < 8; ++i) am[i] = 0;
    }

    for (int c = 1; c < NUM_CLASSES; ++c) {
        const size_t o  = (size_t)c * HW4;
        const float4 c0 = pp[o];
        const float4 c1 = pp[o + 1];
        float v[8];
        v[0] = c0.x; v[1] = c0.y; v[2] = c0.z; v[3] = c0.w;
        v[4] = c1.x; v[5] = c1.y; v[6] = c1.z; v[7] = c1.w;
#pragma unroll
        for (int i = 0; i < 8; ++i)
            if (v[i] > mx[i]) { mx[i] = v[i]; am[i] = c; }   // strict > : first max wins
    }

    // 8 int64 labels as 4x b128 loads (low dword of each int64 carries the value,
    // including the sign bits for IGNORE = -1).
    int l[8];
    {
        const int4* __restrict__ lab4 = (const int4*)label + (p0 >> 1);
#pragma unroll
        for (int j = 0; j < 4; ++j) {
            const int4 L = lab4[j];
            l[2 * j]     = L.x;
            l[2 * j + 1] = L.z;
        }
    }

#pragma unroll
    for (int i = 0; i < 8; ++i) {
        const int li = l[i];
        if (li != -1) {
            int lc = li;
            if (lc < 0) lc = 0;
            if (lc > NUM_CLASSES - 1) lc = NUM_CLASSES - 1;
            if (am[i] == li) atomicAdd(&hist[0 * 32 + lc], 1u);  // intersection
            atomicAdd(&hist[1 * 32 + am[i]], 1u);                // pred area
            atomicAdd(&hist[2 * 32 + lc], 1u);                   // label area
        }
    }
    __syncthreads();

    // Counts <= 2048 are exactly representable in f16. Padded bins stay 0.
    for (int k = t; k < HIST_STRIDE; k += 256)
        partial[(size_t)blk * HIST_STRIDE + k] = (_Float16)(float)hist[k];
}

// ---------------------------------------------------------------------------
// Stage 2: WMMA reduction. 6 waves: (histogram c in 0..2) x (column group 0..1).
// Each wave sums NBLK=2048 partial rows for its 16 columns, 32 rows per
// v_wmma_f32_16x16x32_f16 via the all-ones-A trick. The K-slot mapping of the
// B operand is irrelevant (sum over K is order-independent); only
// "column = lane % 16" matters, which the ISA B-layout guarantees.
// ---------------------------------------------------------------------------
__global__ __launch_bounds__(192) void iou_reduce_kernel(
    const _Float16* __restrict__ partial,
    float* __restrict__ out)
{
    __shared__ float totals[HIST_STRIDE];

    const int t    = threadIdx.x;
    const int lane = t & 31;
    const int wave = t >> 5;     // 0..5, full wave32s => EXEC all ones for WMMA
    const int c    = wave >> 1;  // histogram id 0..2
    const int grp  = wave & 1;   // bin group: bins [0,16) or [16,32)
    const int col  = lane & 15;  // N = lane % 16
    const int g    = lane >> 4;  // lane group selects half of the 32 K rows

    v16h ones;
#pragma unroll
    for (int i = 0; i < 16; ++i) ones[i] = (_Float16)1.0f;

    v8f acc = {};
    for (int it = 0; it < NBLK / 32; ++it) {
        const int rowbase = it * 32 + g * 16;
        v16h bm;
#pragma unroll
        for (int i = 0; i < 16; ++i)
            bm[i] = partial[(size_t)(rowbase + i) * HIST_STRIDE
                            + c * 32 + grp * 16 + col];
        // D = ones(16x32) * B(32x16) + C  -> every row of D = column sums of B
        acc = __builtin_amdgcn_wmma_f32_16x16x32_f16(
            /*neg_a=*/false, ones, /*neg_b=*/false, bm,
            /*c_mod=*/(short)0, acc, /*reuse_a=*/false, /*reuse_b=*/false);
    }

    // All D rows are identical; lane L (<16) element 0 holds the sum for its column.
    if (lane < 16) totals[c * 32 + grp * 16 + col] = acc[0];
    __syncthreads();

    if (t == 0) {
        float s   = 0.0f;
        int   cnt = 0;
        for (int bin = 0; bin < NUM_CLASSES; ++bin) {
            const float inter = totals[0 * 32 + bin];
            const float ap    = totals[1 * 32 + bin];
            const float al    = totals[2 * 32 + bin];
            const float uni   = ap + al - inter;
            if (uni != 0.0f) { s += inter / uni; ++cnt; }  // 0/0 -> NaN -> skipped
        }
        const float mean = (cnt > 0) ? (s / (float)cnt) : 0.5f;  // all-NaN -> 0.5
        out[0] = 1.0f - mean;
    }
}

// ---------------------------------------------------------------------------
extern "C" void kernel_launch(void* const* d_in, const int* in_sizes, int n_in,
                              void* d_out, int out_size, void* d_ws, size_t ws_size,
                              hipStream_t stream)
{
    (void)in_sizes; (void)n_in; (void)out_size; (void)ws_size;

    const float*     pred  = (const float*)d_in[0];       // [8,19,512,1024] f32
    const long long* label = (const long long*)d_in[1];   // [8,512,1024] i64
    _Float16* partial = (_Float16*)d_ws;                  // NBLK * 96 f16 = 384 KiB
    float*    out     = (float*)d_out;

    iou_hist_kernel<<<NBLK, 256, 0, stream>>>(pred, label, partial);
    iou_reduce_kernel<<<1, 192, 0, stream>>>(partial, out);
}
